// SGNN_59356448031280
// MI455X (gfx1250) — matrix operand order repaired and checked by reading, not compiled, not run on hardware
//
#include <hip/hip_runtime.h>
#include <math.h>

// ---------------------------------------------------------------------------
// Types for CDNA5 WMMA
// ---------------------------------------------------------------------------
typedef __attribute__((ext_vector_type(16))) __bf16 v16bf;
typedef __attribute__((ext_vector_type(8)))  float  v8f;

union FragBF {
    v16bf    v;
    unsigned u[8];
};

__device__ __forceinline__ unsigned short f2bf(float f) {
    unsigned u = __builtin_bit_cast(unsigned, f);
    u += 0x7fffu + ((u >> 16) & 1u);   // round-to-nearest-even
    return (unsigned short)(u >> 16);
}

// CDNA5 async global->LDS copy (ASYNCcnt-tracked), ISA §15.18 opcode 98.
// lds_byte_off: LDS byte address (low 32 bits of a generic shared pointer).
__device__ __forceinline__ void async_ld_b128(unsigned lds_byte_off, const void* gaddr) {
    asm volatile("global_load_async_to_lds_b128 %0, %1, off"
                 :: "v"(lds_byte_off), "v"(gaddr)
                 : "memory");
}
__device__ __forceinline__ void wait_async0() {
    asm volatile("s_wait_asynccnt 0x0" ::: "memory");
}

// ---------------------------------------------------------------------------
// Problem constants
// ---------------------------------------------------------------------------
#define Bsz   4096
#define Hdim  512
#define Rows  13
#define Mgru  (Bsz * Rows)          // 53248
#define HALF  256                    // H/2

// ---------------------------------------------------------------------------
// f32 -> bf16 convert (weights)
// ---------------------------------------------------------------------------
__global__ void k_f2bf(const float* __restrict__ in, unsigned short* __restrict__ out, int n) {
    int i = blockIdx.x * blockDim.x + threadIdx.x;
    if (i < n) out[i] = f2bf(in[i]);
}

// ---------------------------------------------------------------------------
// Embedding gather + 4-way feature concat:
// h[b, r, seg*128 + e] = emb[ ids[b, seg*13 + r], e ]
// ---------------------------------------------------------------------------
__global__ __launch_bounds__(256) void k_embed(const int* __restrict__ ids,
                                               const float* __restrict__ emb,
                                               float* __restrict__ hf,
                                               unsigned short* __restrict__ hb) {
    size_t i = (size_t)blockIdx.x * blockDim.x + threadIdx.x;  // B*13*512
    int e   = (int)(i & 127);
    int seg = (int)((i >> 7) & 3);
    int r   = (int)((i >> 9) % Rows);
    int b   = (int)(i / (Rows * Hdim));
    int tok = ids[(size_t)b * 52 + seg * Rows + r];
    float v = emb[(size_t)tok * 128 + e];
    hf[i] = v;
    hb[i] = f2bf(v);
}

// ---------------------------------------------------------------------------
// a = At @ h + b_ah   (batched 13x13 @ 13x512), output bf16.
// One block per (b,i) output row.
// ---------------------------------------------------------------------------
__global__ __launch_bounds__(256) void k_batched_a(const float* __restrict__ Adat,
                                                   const float* __restrict__ hf,
                                                   const float* __restrict__ b_ah,
                                                   unsigned short* __restrict__ abf) {
    int bi = blockIdx.x;              // b*13 + i
    int b  = bi / Rows;
    int i  = bi % Rows;
    const float* Ab    = Adat + (size_t)b * (Rows * Rows);
    const float* hbase = hf + (size_t)b * Rows * Hdim;
    float areg[Rows];
#pragma unroll
    for (int j = 0; j < Rows; ++j) areg[j] = Ab[j * Rows + i];   // At[i][j] = A[j][i]
    for (int n = threadIdx.x; n < Hdim; n += 256) {
        float s = b_ah[n];
#pragma unroll
        for (int j = 0; j < Rows; ++j) s = fmaf(areg[j], hbase[j * Hdim + n], s);
        abf[(size_t)bi * Hdim + n] = f2bf(s);
    }
}

// ---------------------------------------------------------------------------
// WMMA GEMM:  C[M,N] (f32) = A[M,K] (bf16) * W[N,K]^T (bf16)
//   Block tile 128x64, wave tile 32x32 (4 WMMA/step), K-step 32.
//   Double-buffered LDS fed by async global->LDS copies (ASYNCcnt).
//   accumulate: C += ...     bias/relu applied at final store.
// ---------------------------------------------------------------------------
#define BM 128
#define BN 64
#define BK 32
#define LSTR   40                 // padded LDS row stride (elems); rows 16B aligned
#define ASTAGE (BM * LSTR)        // 5120 elems / stage
#define BSTAGE (BN * LSTR)        // 2560 elems / stage

__global__ __launch_bounds__(256) void k_gemm_bf16(const unsigned short* __restrict__ A,
                                                   const unsigned short* __restrict__ W,
                                                   const float* __restrict__ bias,
                                                   float* __restrict__ C,
                                                   int M, int N, int K,
                                                   int accumulate, int relu) {
    __shared__ unsigned short sA[2 * ASTAGE];
    __shared__ unsigned short sB[2 * BSTAGE];

    const int tid  = threadIdx.x;
    const int wave = tid >> 5;
    const int lane = tid & 31;
    const int m0   = blockIdx.x * BM;
    const int n0   = blockIdx.y * BN;

    const int wm = wave & 3;            // 4 row tiles of 32
    const int wn = wave >> 2;           // 2 col tiles of 32

    v8f acc00 = {}, acc01 = {}, acc10 = {}, acc11 = {};

    // C/D layout: element v of lane l -> row += (l>=16)*8 + v, col = (l&15)
    const int rbase = wm * 32 + (lane >> 4) * 8;    // + v, +16 for A-tile 1
    const int cbase = wn * 32 + (lane & 15);        // +16 for B-tile 1

    if (accumulate) {
#pragma unroll
        for (int v = 0; v < 8; ++v) {
            size_t r0 = (size_t)(m0 + rbase + v) * N;
            size_t r1 = (size_t)(m0 + rbase + 16 + v) * N;
            acc00[v] = C[r0 + n0 + cbase];
            acc01[v] = C[r0 + n0 + cbase + 16];
            acc10[v] = C[r1 + n0 + cbase];
            acc11[v] = C[r1 + n0 + cbase + 16];
        }
    }

    // fragment addressing (per documented CDNA5 VGPR layouts)
    const int amBase = wm * 32 + (lane & 15);
    const int kh     = (lane >> 4) * 8;     // A: lane half -> K offset 0/8
    const int bnBase = wn * 32 + (lane & 15);
    const int kb     = (lane >> 4) * 8;     // B: lane half -> K offset 0/8

    // async stage loader: A tile 128x32 (512 x b128), B tile 64x32 (256 x b128)
    auto issue_stage = [&](int k0, int st) {
        int q, row, seg;
        q = tid;            row = q >> 2; seg = (q & 3) * 8;
        async_ld_b128((unsigned)(size_t)&sA[st * ASTAGE + row * LSTR + seg],
                      A + (size_t)(m0 + row) * K + k0 + seg);
        q = tid + 256;      row = q >> 2; seg = (q & 3) * 8;
        async_ld_b128((unsigned)(size_t)&sA[st * ASTAGE + row * LSTR + seg],
                      A + (size_t)(m0 + row) * K + k0 + seg);
        q = tid;            row = q >> 2; seg = (q & 3) * 8;
        async_ld_b128((unsigned)(size_t)&sB[st * BSTAGE + row * LSTR + seg],
                      W + (size_t)(n0 + row) * K + k0 + seg);
    };

    issue_stage(0, 0);
    wait_async0();
    __syncthreads();

    const int nsteps = K / BK;
    for (int s = 0; s < nsteps; ++s) {
        const int cur = s & 1;
        if (s + 1 < nsteps) issue_stage((s + 1) * BK, cur ^ 1);

        const unsigned short* aB = &sA[cur * ASTAGE];
        const unsigned short* bB = &sB[cur * BSTAGE];
        const unsigned short* pa0 = aB + (size_t)amBase * LSTR;
        const unsigned short* pa1 = aB + (size_t)(amBase + 16) * LSTR;
        const unsigned short* pb0 = bB + (size_t)bnBase * LSTR + kb;
        const unsigned short* pb1 = bB + (size_t)(bnBase + 16) * LSTR + kb;

        FragBF fa0, fa1, fb0, fb1;
#pragma unroll
        for (int ep = 0; ep < 8; ++ep) {
            int e = ep * 2;
            int k = ((e & 8) << 1) + (e & 7) + kh;      // A K-striping
            fa0.u[ep] = *(const unsigned*)&pa0[k];
            fa1.u[ep] = *(const unsigned*)&pa1[k];
            fb0.u[ep] = *(const unsigned*)&pb0[k - kh]; // B K-striping (same as A)
            fb1.u[ep] = *(const unsigned*)&pb1[k - kh];
        }

        acc00 = __builtin_amdgcn_wmma_f32_16x16x32_bf16(false, fa0.v, false, fb0.v,
                                                        (short)0, acc00, false, false);
        acc01 = __builtin_amdgcn_wmma_f32_16x16x32_bf16(false, fa0.v, false, fb1.v,
                                                        (short)0, acc01, false, false);
        acc10 = __builtin_amdgcn_wmma_f32_16x16x32_bf16(false, fa1.v, false, fb0.v,
                                                        (short)0, acc10, false, false);
        acc11 = __builtin_amdgcn_wmma_f32_16x16x32_bf16(false, fa1.v, false, fb1.v,
                                                        (short)0, acc11, false, false);

        wait_async0();       // next stage resident before anyone reads it
        __syncthreads();     // and all waves done reading current stage
    }

#pragma unroll
    for (int v = 0; v < 8; ++v) {
        size_t r0 = (size_t)(m0 + rbase + v) * N;
        size_t r1 = (size_t)(m0 + rbase + 16 + v) * N;
        float x00 = acc00[v], x01 = acc01[v], x10 = acc10[v], x11 = acc11[v];
        if (bias) {
            float b0 = bias[n0 + cbase], b1 = bias[n0 + cbase + 16];
            x00 += b0; x01 += b1; x10 += b0; x11 += b1;
        }
        if (relu) {
            x00 = fmaxf(x00, 0.f); x01 = fmaxf(x01, 0.f);
            x10 = fmaxf(x10, 0.f); x11 = fmaxf(x11, 0.f);
        }
        C[r0 + n0 + cbase]      = x00;
        C[r0 + n0 + cbase + 16] = x01;
        C[r1 + n0 + cbase]      = x10;
        C[r1 + n0 + cbase + 16] = x11;
    }
}

// ---------------------------------------------------------------------------
// GRU elementwise stage 1: z = sigmoid(zpre + b), r = sigmoid(rpre + b),
// rh = bf16(r * h). z overwrites zpre.
// ---------------------------------------------------------------------------
__global__ __launch_bounds__(256) void k_ew1(float* __restrict__ zpre,
                                             const float* __restrict__ rpre,
                                             const float* __restrict__ hf,
                                             const float* __restrict__ b_wz,
                                             const float* __restrict__ b_uz,
                                             const float* __restrict__ b_wr,
                                             const float* __restrict__ b_ur,
                                             unsigned short* __restrict__ rhbf) {
    size_t i = (size_t)blockIdx.x * blockDim.x + threadIdx.x;  // Mgru*512
    int n = (int)(i & (Hdim - 1));
    float z = 1.f / (1.f + __expf(-(zpre[i] + b_wz[n] + b_uz[n])));
    float r = 1.f / (1.f + __expf(-(rpre[i] + b_wr[n] + b_ur[n])));
    zpre[i] = z;
    rhbf[i] = f2bf(r * hf[i]);
}

// ---------------------------------------------------------------------------
// GRU elementwise stage 2: c = tanh(aw + b_w + rhu + b_u);
// h = (1-z)*h + z*c  (update fp32 + bf16 mirror)
// ---------------------------------------------------------------------------
__global__ __launch_bounds__(256) void k_ew2(const float* __restrict__ z,
                                             const float* __restrict__ aw,
                                             const float* __restrict__ rhu,
                                             const float* __restrict__ b_w,
                                             const float* __restrict__ b_u,
                                             float* __restrict__ hf,
                                             unsigned short* __restrict__ hbf) {
    size_t i = (size_t)blockIdx.x * blockDim.x + threadIdx.x;
    int n = (int)(i & (Hdim - 1));
    float c  = tanhf(aw[i] + b_w[n] + rhu[i] + b_u[n]);
    float zz = z[i];
    float hn = (1.f - zz) * hf[i] + zz * c;
    hf[i]  = hn;
    hbf[i] = f2bf(hn);
}

// ---------------------------------------------------------------------------
// Pack strided rows of h (bf16) into dense matrices for the attention GEMMs.
// out[(b*rows_per_b + j), :] = hbf[(b*13 + row_off + j), :]
// ---------------------------------------------------------------------------
__global__ __launch_bounds__(256) void k_pack(const unsigned short* __restrict__ hbf,
                                              unsigned short* __restrict__ out,
                                              int rows_per_b, int row_off) {
    size_t i = (size_t)blockIdx.x * blockDim.x + threadIdx.x;  // B*rows_per_b*512
    int d = (int)(i & (Hdim - 1));
    int j = (int)((i >> 9) % rows_per_b);
    int b = (int)(i / ((size_t)rows_per_b * Hdim));
    out[i] = hbf[((size_t)(b * Rows + row_off + j)) * Hdim + d];
}

// ---------------------------------------------------------------------------
// Row-dot projection: out[row] = relu( dot(X[row,0:256], wv) + bias ).
// One wave (32 lanes) per row.
// ---------------------------------------------------------------------------
__global__ __launch_bounds__(256) void k_rowdot(const float* __restrict__ X,
                                                const float* __restrict__ wv,
                                                const float* __restrict__ bsc,
                                                float* __restrict__ out, int M) {
    int row  = (int)((blockIdx.x * (size_t)blockDim.x + threadIdx.x) >> 5);
    int lane = threadIdx.x & 31;
    if (row >= M) return;
    const float* x = X + (size_t)row * HALF;
    float s = 0.f;
#pragma unroll
    for (int e = lane; e < HALF; e += 32) s += x[e] * wv[e];
#pragma unroll
    for (int off = 16; off; off >>= 1) s += __shfl_xor(s, off, 32);
    if (lane == 0) out[row] = fmaxf(s + bsc[0], 0.f);
}

// ---------------------------------------------------------------------------
// Final scoring: per (b,k): softmax-like weights over j=0..7, weighted sum of
// ctx rows, Euclidean distance to input_b row, score = -||.||.
// One wave per (b,k).
// ---------------------------------------------------------------------------
__global__ __launch_bounds__(256) void k_final(const float* __restrict__ ua2,
                                               const float* __restrict__ ub2,
                                               const float* __restrict__ hf,
                                               float* __restrict__ out) {
    int idx  = (int)((blockIdx.x * (size_t)blockDim.x + threadIdx.x) >> 5);  // b*5+k
    int lane = threadIdx.x & 31;
    if (idx >= Bsz * 5) return;
    int b = idx / 5, k = idx % 5;

    const float* ua = ua2 + (size_t)b * 8;
    float ub = ub2[idx];
    float wj[8];
    float wsum = 0.f;
#pragma unroll
    for (int j = 0; j < 8; ++j) {
        float t = __expf(tanhf(ua[j] + ub));
        wj[j] = t;
        wsum += t;
    }
    float inv = 1.f / wsum;

    const float* ctx = hf + (size_t)b * Rows * Hdim;
    const float* ib  = ctx + (size_t)(8 + k) * Hdim;
    float sq = 0.f;
    for (int d = lane; d < Hdim; d += 32) {
        float av = 0.f;
#pragma unroll
        for (int j = 0; j < 8; ++j) av = fmaf(ctx[j * Hdim + d], wj[j], av);
        float diff = av * inv - ib[d];
        sq = fmaf(diff, diff, sq);
    }
#pragma unroll
    for (int off = 16; off; off >>= 1) sq += __shfl_xor(sq, off, 32);
    if (lane == 0) out[idx] = -sqrtf(sq);
}

// ---------------------------------------------------------------------------
// Host orchestration
// ---------------------------------------------------------------------------
static inline void launch_gemm(const unsigned short* A, const unsigned short* W,
                               const float* bias, float* C, int M, int N, int K,
                               int acc, int relu, hipStream_t s) {
    dim3 grid(M / BM, N / BN), block(256);
    k_gemm_bf16<<<grid, block, 0, s>>>(A, W, bias, C, M, N, K, acc, relu);
}

extern "C" void kernel_launch(void* const* d_in, const int* in_sizes, int n_in,
                              void* d_out, int out_size, void* d_ws, size_t ws_size,
                              hipStream_t stream) {
    (void)in_sizes; (void)n_in; (void)out_size; (void)ws_size;

    const float* A_in   = (const float*)d_in[0];
    const int*   ids    = (const int*)d_in[1];
    const float* emb    = (const float*)d_in[2];
    const float* b_ah   = (const float*)d_in[3];
    const float* w_z    = (const float*)d_in[4];
    const float* b_wz   = (const float*)d_in[5];
    const float* u_z    = (const float*)d_in[6];
    const float* b_uz   = (const float*)d_in[7];
    const float* w_r    = (const float*)d_in[8];
    const float* b_wr   = (const float*)d_in[9];
    const float* u_r    = (const float*)d_in[10];
    const float* b_ur   = (const float*)d_in[11];
    const float* w_     = (const float*)d_in[12];
    const float* b_w    = (const float*)d_in[13];
    const float* u_m    = (const float*)d_in[14];
    const float* b_u    = (const float*)d_in[15];
    const float* w1     = (const float*)d_in[16];
    const float* b1     = (const float*)d_in[17];
    const float* w12    = (const float*)d_in[18];
    const float* b12    = (const float*)d_in[19];
    const float* w2     = (const float*)d_in[20];
    const float* b2     = (const float*)d_in[21];
    const float* w22    = (const float*)d_in[22];
    const float* b22    = (const float*)d_in[23];

    // ---- workspace carve-out ----
    char* ws = (char*)d_ws;
    auto carve = [&](size_t bytes) -> void* {
        void* p = ws;
        ws += (bytes + 255) & ~(size_t)255;
        return p;
    };
    const size_t HH = (size_t)Hdim * Hdim;            // 262144
    unsigned short* wz_bf = (unsigned short*)carve(HH * 2);
    unsigned short* uz_bf = (unsigned short*)carve(HH * 2);
    unsigned short* wr_bf = (unsigned short*)carve(HH * 2);
    unsigned short* ur_bf = (unsigned short*)carve(HH * 2);
    unsigned short* w_bf  = (unsigned short*)carve(HH * 2);
    unsigned short* um_bf = (unsigned short*)carve(HH * 2);
    unsigned short* w1_bf = (unsigned short*)carve((size_t)HALF * Hdim * 2);
    unsigned short* w2_bf = (unsigned short*)carve((size_t)HALF * Hdim * 2);

    const size_t MH = (size_t)Mgru * Hdim;            // 53248*512
    float*          h_f32 = (float*)carve(MH * 4);
    unsigned short* h_bf  = (unsigned short*)carve(MH * 2);
    unsigned short* a_bf  = (unsigned short*)carve(MH * 2);   // reused as ctx_bf
    float*          zpre  = (float*)carve(MH * 4);            // reused as h1a
    float*          rpre  = (float*)carve(MH * 4);            // reused as rhu, then ua2/ub2
    float*          awpre = (float*)carve(MH * 4);            // reused as h1b
    unsigned short* rh_bf = (unsigned short*)carve(MH * 2);   // reused as hb_bf

    // ---- convert weights to bf16 ----
    {
        dim3 b(256);
        k_f2bf<<<dim3((HH + 255) / 256), b, 0, stream>>>(w_z, wz_bf, (int)HH);
        k_f2bf<<<dim3((HH + 255) / 256), b, 0, stream>>>(u_z, uz_bf, (int)HH);
        k_f2bf<<<dim3((HH + 255) / 256), b, 0, stream>>>(w_r, wr_bf, (int)HH);
        k_f2bf<<<dim3((HH + 255) / 256), b, 0, stream>>>(u_r, ur_bf, (int)HH);
        k_f2bf<<<dim3((HH + 255) / 256), b, 0, stream>>>(w_,  w_bf,  (int)HH);
        k_f2bf<<<dim3((HH + 255) / 256), b, 0, stream>>>(u_m, um_bf, (int)HH);
        size_t hw = (size_t)HALF * Hdim;
        k_f2bf<<<dim3((hw + 255) / 256), b, 0, stream>>>(w1, w1_bf, (int)hw);
        k_f2bf<<<dim3((hw + 255) / 256), b, 0, stream>>>(w2, w2_bf, (int)hw);
    }

    // ---- embedding gather ----
    k_embed<<<dim3((unsigned)(MH / 256)), dim3(256), 0, stream>>>(ids, emb, h_f32, h_bf);

    const unsigned ewBlocks = (unsigned)(MH / 256);

    // ---- GRU iterations ----
    for (int it = 0; it < 2; ++it) {
        k_batched_a<<<dim3(Mgru), dim3(256), 0, stream>>>(A_in, h_f32, b_ah, a_bf);

        launch_gemm(a_bf, wz_bf, nullptr, zpre,  Mgru, Hdim, Hdim, 0, 0, stream);
        launch_gemm(h_bf, uz_bf, nullptr, zpre,  Mgru, Hdim, Hdim, 1, 0, stream);
        launch_gemm(a_bf, wr_bf, nullptr, rpre,  Mgru, Hdim, Hdim, 0, 0, stream);
        launch_gemm(h_bf, ur_bf, nullptr, rpre,  Mgru, Hdim, Hdim, 1, 0, stream);
        launch_gemm(a_bf, w_bf,  nullptr, awpre, Mgru, Hdim, Hdim, 0, 0, stream);

        k_ew1<<<dim3(ewBlocks), dim3(256), 0, stream>>>(zpre, rpre, h_f32,
                                                        b_wz, b_uz, b_wr, b_ur, rh_bf);

        launch_gemm(rh_bf, um_bf, nullptr, rpre, Mgru, Hdim, Hdim, 0, 0, stream);  // rhu

        k_ew2<<<dim3(ewBlocks), dim3(256), 0, stream>>>(zpre, awpre, rpre,
                                                        b_w, b_u, h_f32, h_bf);
    }

    // ---- attention / scoring head ----
    const int Mctx = Bsz * 8;    // 32768
    const int Mhb  = Bsz * 5;    // 20480
    unsigned short* ctx_bf = a_bf;    // reuse
    unsigned short* hb_bf  = rh_bf;   // reuse
    float* h1a = zpre;                // reuse (32768*256 f32)
    float* h1b = awpre;               // reuse (20480*256 f32)
    float* ua2 = rpre;                // reuse (32768 f32)
    float* ub2 = rpre + Mctx;         // (20480 f32)

    k_pack<<<dim3((unsigned)(((size_t)Mctx * Hdim) / 256)), dim3(256), 0, stream>>>(h_bf, ctx_bf, 8, 0);
    k_pack<<<dim3((unsigned)(((size_t)Mhb  * Hdim) / 256)), dim3(256), 0, stream>>>(h_bf, hb_bf,  5, 8);

    launch_gemm(ctx_bf, w1_bf, b1, h1a, Mctx, HALF, Hdim, 0, 1, stream);  // relu(ctx@w1^T+b1)
    launch_gemm(hb_bf,  w2_bf, b2, h1b, Mhb,  HALF, Hdim, 0, 1, stream);  // relu(hb@w2^T+b2)

    k_rowdot<<<dim3(Mctx / 8), dim3(256), 0, stream>>>(h1a, w12, b12, ua2, Mctx);
    k_rowdot<<<dim3(Mhb  / 8), dim3(256), 0, stream>>>(h1b, w22, b22, ub2, Mhb);

    k_final<<<dim3(Mhb / 8), dim3(256), 0, stream>>>(ua2, ub2, h_f32, (float*)d_out);
}
